// MultiheadDiff2_87600152969380
// MI455X (gfx1250) — matrix operand order repaired and checked by reading, not compiled, not run on hardware
//
#include <hip/hip_runtime.h>
#include <hip/hip_bf16.h>
#include <math.h>

#define T_SEQ 2048
#define EMB   2048
#define NHEAD 16
#define HD    64
#define HD2   128
#define LAMBDA_INIT_F  0.7836057665f   // 0.8 - 0.6*exp(-0.3*12)
#define ONE_MINUS_LI   0.2163942335f
#define EPS_RMS        1e-5f

typedef __attribute__((ext_vector_type(16))) __bf16 v16bf;
typedef __attribute__((ext_vector_type(8)))  __bf16 v8bf;
typedef __attribute__((ext_vector_type(8)))  float  v8f;
typedef __attribute__((ext_vector_type(4)))  unsigned int v4u;
typedef __attribute__((ext_vector_type(8)))  int v8i;
typedef __attribute__((ext_vector_type(4)))  int v4i;

// --- CDNA5 feature detection (device pass only; host pass takes fallbacks) ---
#if defined(__HIP_DEVICE_COMPILE__)
#if __has_builtin(__builtin_amdgcn_tensor_load_to_lds) && __has_builtin(__builtin_amdgcn_s_wait_tensorcnt)
#define HAVE_TDM 1
#endif
#if __has_builtin(__builtin_amdgcn_global_load_async_to_lds_b128) && __has_builtin(__builtin_amdgcn_s_wait_asynccnt)
#define HAVE_ASYNC 1
#endif
#endif

#define AS1 __attribute__((address_space(1)))
#define AS3 __attribute__((address_space(3)))

__device__ __forceinline__ v16bf pack16(v8bf lo, v8bf hi) {
    return __builtin_shufflevector(lo, hi, 0,1,2,3,4,5,6,7,8,9,10,11,12,13,14,15);
}
__device__ __forceinline__ v16bf ld_frag(const __bf16* p0, const __bf16* p1) {
    v8bf lo = *(const v8bf*)p0;
    v8bf hi = *(const v8bf*)p1;
    return pack16(lo, hi);
}
__device__ __forceinline__ v8bf cvt8(const float* s) {
    v8bf r;
#pragma unroll
    for (int i = 0; i < 8; ++i) r[i] = (__bf16)s[i];
    return r;
}

#if defined(HAVE_TDM)
// Issue one TDM 2D-tile load (count=1 descriptor, type=2).
__device__ __forceinline__ void tdm_load_2d(unsigned lds_addr, unsigned long long gaddr,
                                            const v8i& g1) {
    v4i gz = {0, 0, 0, 0};
    v4u g0;
    g0[0] = 1u;                                            // count=1
    g0[1] = lds_addr;                                      // lds_addr
    g0[2] = (unsigned)(gaddr & 0xFFFFFFFFu);               // global_addr lo
    g0[3] = (unsigned)((gaddr >> 32) & 0x01FFFFFFu) | (2u << 30); // hi | type=2
#if __clang_major__ >= 23
    v8i gx = {0, 0, 0, 0, 0, 0, 0, 0};
    __builtin_amdgcn_tensor_load_to_lds(g0, g1, gz, gz, gx, 0);
#else
    __builtin_amdgcn_tensor_load_to_lds(g0, g1, gz, gz, 0);
#endif
}
#endif

// ---------------------------------------------------------------------------
// C[M,N] = A[M,K] * B[N,K]^T (NT GEMM). Block tile 128x128, 8 waves each
// 32x64, K-step 64 => 16 WMMA per iteration. Software pipelined:
// double-buffered LDS (single barrier / iter), next-tile loads issued before
// the WMMA work. When A is bf16 and TDM is available, the A tile is DMA'd by
// the Tensor Data Mover into the alternate buffer, overlapped with compute.
// ---------------------------------------------------------------------------
template <typename TA, typename TOUT>
__global__ __launch_bounds__(256) void gemm_nt(const TA* __restrict__ A,
                                               const float* __restrict__ Bw,
                                               TOUT* __restrict__ C,
                                               int M, int N, int K) {
    __shared__ __align__(16) __bf16 As[2][128][64];
    __shared__ __align__(16) __bf16 Bs[2][128][64];

    const int tid  = threadIdx.x;
    const int lane = tid & 31;
    const int w    = tid >> 5;
    const int wm   = w & 3;
    const int wn   = w >> 2;
    const int mBase = blockIdx.y * 128;
    const int nBase = blockIdx.x * 128;

    v8f acc[2][4];
#pragma unroll
    for (int i = 0; i < 2; ++i)
#pragma unroll
        for (int j = 0; j < 4; ++j)
#pragma unroll
            for (int e = 0; e < 8; ++e) acc[i][j][e] = 0.f;

    const int lr = tid >> 1;            // 0..127 tile row for loads
    const int lk = (tid & 1) * 32;      // 0 or 32

    // ---- B register staging (weights fp32 -> bf16 at store time) ----
    const float* bPtr = Bw + (size_t)(nBase + lr) * K + lk;
    float bF[32];
#pragma unroll
    for (int i = 0; i < 32; ++i) bF[i] = bPtr[i];

    // ---- A staging state ----
    float aF[32];
    v8bf  aB[4];
#if defined(HAVE_TDM)
    v8i g1t;
    const unsigned ldsA0 = (unsigned)(uintptr_t)(void*)&As[0][0][0];
    if constexpr (sizeof(TA) == 2) {
        g1t[0] = (1 << 16);           // data_size=2B, no padding
        g1t[1] = (int)(64u  << 16);   // tensor_dim0 = 64
        g1t[2] = (int)(128u << 16);   // tensor_dim0 hi=0 ; tensor_dim1 = 128
        g1t[3] = (int)(64u  << 16);   // tensor_dim1 hi=0 ; tile_dim0 = 64
        g1t[4] = 128;                 // tile_dim1 = 128 ; tile_dim2 = 0
        g1t[5] = K;                   // tensor_dim0_stride (elements)
        g1t[6] = 0;
        g1t[7] = 0;
        if (w == 0)                   // prologue DMA: tile 0 -> As[0]
            tdm_load_2d(ldsA0,
                (unsigned long long)(uintptr_t)((const __bf16*)A + (size_t)mBase * K),
                g1t);
    } else
#endif
    {
        if constexpr (sizeof(TA) == 4) {
            const float* ap = (const float*)A + (size_t)(mBase + lr) * K + lk;
#pragma unroll
            for (int i = 0; i < 32; ++i) aF[i] = ap[i];
        } else {
            const __bf16* ap = (const __bf16*)A + (size_t)(mBase + lr) * K + lk;
#pragma unroll
            for (int i = 0; i < 4; ++i) aB[i] = *(const v8bf*)(ap + i * 8);
        }
    }

    int buf = 0;
    for (int kt = 0; kt < K; kt += 64, buf ^= 1) {
        const bool hasNext = (kt + 64 < K);

        // ---- commit staged tile to LDS[buf] ----
#pragma unroll
        for (int i = 0; i < 4; ++i)
            *(v8bf*)&Bs[buf][lr][lk + i * 8] = cvt8(bF + i * 8);
#if defined(HAVE_TDM)
        if constexpr (sizeof(TA) == 2) {
            if (w == 0) __builtin_amdgcn_s_wait_tensorcnt(0);  // As[buf] DMA done
        } else
#endif
        {
            if constexpr (sizeof(TA) == 4) {
#pragma unroll
                for (int i = 0; i < 4; ++i)
                    *(v8bf*)&As[buf][lr][lk + i * 8] = cvt8(aF + i * 8);
            } else {
#pragma unroll
                for (int i = 0; i < 4; ++i)
                    *(v8bf*)&As[buf][lr][lk + i * 8] = aB[i];
            }
        }
        __syncthreads();   // single barrier per iteration (ping-pong buffers)

        // ---- kick off next tile; overlaps with the WMMA work below ----
        if (hasNext) {
#if defined(HAVE_TDM)
            if constexpr (sizeof(TA) == 2) {
                if (w == 0)
                    tdm_load_2d(ldsA0 + (unsigned)((buf ^ 1) * 128 * 64 * 2),
                        (unsigned long long)(uintptr_t)
                            ((const __bf16*)A + (size_t)mBase * K + kt + 64),
                        g1t);
            } else
#endif
            {
                if constexpr (sizeof(TA) == 4) {
                    const float* ap = (const float*)A + (size_t)(mBase + lr) * K + kt + 64 + lk;
#pragma unroll
                    for (int i = 0; i < 32; ++i) aF[i] = ap[i];
                } else {
                    const __bf16* ap = (const __bf16*)A + (size_t)(mBase + lr) * K + kt + 64 + lk;
#pragma unroll
                    for (int i = 0; i < 4; ++i) aB[i] = *(const v8bf*)(ap + i * 8);
                }
            }
            const float* bp = bPtr + kt + 64;
            __builtin_prefetch(bp + 64, 0, 1);        // next-next B tile -> L2
#pragma unroll
            for (int i = 0; i < 32; ++i) bF[i] = bp[i];
        }

        // ---- 16 WMMA from LDS[buf] ----
        const int kbase = (lane < 16) ? 0 : 8;
        const int koff  = (lane < 16) ? 0 : 8;
#pragma unroll
        for (int kk = 0; kk < 2; ++kk) {
            const int kO = kk * 32;
            v16bf af[2];
#pragma unroll
            for (int i = 0; i < 2; ++i) {
                const int r0 = wm * 32 + i * 16 + (lane & 15);
                af[i] = ld_frag(&As[buf][r0][kO + kbase], &As[buf][r0][kO + kbase + 16]);
            }
#pragma unroll
            for (int j = 0; j < 4; ++j) {
                const int n0 = wn * 64 + j * 16 + (lane & 15);
                v16bf bfr = ld_frag(&Bs[buf][n0][kO + koff], &Bs[buf][n0][kO + koff + 16]);
#pragma unroll
                for (int i = 0; i < 2; ++i)
                    acc[i][j] = __builtin_amdgcn_wmma_f32_16x16x32_bf16(
                        false, af[i], false, bfr, (short)0, acc[i][j], false, false);
            }
        }
        // no trailing barrier: next iteration writes the other buffer
    }

    const int col  = lane & 15;
    const int rOff = (lane >> 4) * 8;
#pragma unroll
    for (int i = 0; i < 2; ++i)
#pragma unroll
        for (int j = 0; j < 4; ++j) {
            const int gm = mBase + wm * 32 + i * 16 + rOff;
            const int gn = nBase + wn * 64 + j * 16 + col;
#pragma unroll
            for (int r = 0; r < 8; ++r) {
                const float vv = acc[i][j][r];
                if constexpr (sizeof(TOUT) == 4) {
                    C[(size_t)(gm + r) * N + gn] = vv;
                } else {
                    C[(size_t)(gm + r) * N + gn] = (TOUT)vv;
                }
            }
        }
}

// ---------------------------------------------------------------------------
__global__ void lambda_kernel(const float* __restrict__ lq1, const float* __restrict__ lk1,
                              const float* __restrict__ lq2, const float* __restrict__ lk2,
                              float* __restrict__ outp) {
    const int L = threadIdx.x;     // 32 threads, hd=64 -> 2 per lane
    float s1 = lq1[L] * lk1[L] + lq1[L + 32] * lk1[L + 32];
    float s2 = lq2[L] * lk2[L] + lq2[L + 32] * lk2[L + 32];
#pragma unroll
    for (int m = 1; m < 32; m <<= 1) {
        s1 += __shfl_xor(s1, m, 32);
        s2 += __shfl_xor(s2, m, 32);
    }
    if (L == 0) *outp = __expf(s1) - __expf(s2) + LAMBDA_INIT_F;
}

// ---------------------------------------------------------------------------
// Fused dual flash-attention + diff combine + sub-LN (RMS).
// grid = (T/64, H, B), block 256 (8 waves): waves 0-3 comp 1, 4-7 comp 2.
// K tile staged by TDM (HW row padding -> 136-elt LDS stride) when available.
// ---------------------------------------------------------------------------
__global__ __launch_bounds__(256) void diff_attn(const __bf16* __restrict__ q,
                                                 const __bf16* __restrict__ k,
                                                 const __bf16* __restrict__ v,
                                                 const float* __restrict__ lam_p,
                                                 const float* __restrict__ subln,
                                                 __bf16* __restrict__ o) {
    // region 1 (s-loop): Ks 64x136 (17408B) | Vt 128x72 (18432B) | Ps 8x16x72 (18432B)
    // region 2 (epilogue, unioned): O2s 4x16x128 f32 (32768B)
    __shared__ __align__(16) char smem[54272];
    __bf16* Ks  = (__bf16*)smem;
    __bf16* Vt  = (__bf16*)(smem + 17408);
    __bf16* Ps  = (__bf16*)(smem + 35840);
    float*  O2s = (float*)smem;

    const int tid  = threadIdx.x;
    const int lane = tid & 31;
    const int w    = tid >> 5;
    const int a    = w >> 2;
    const int ms   = w & 3;
    const int col  = lane & 15;
    const int rOff = (lane >> 4) * 8;
    const int b    = blockIdx.z;
    const int h    = blockIdx.y;
    const int m0   = blockIdx.x * 64;

    const float lam   = *lam_p;
    const float scale = 0.125f;       // 1/sqrt(64)
    const int   kbase = (lane < 16) ? 0 : 8;

    v16bf qa[2];
    {
        const __bf16* qrow = q + (size_t)(b * T_SEQ + m0 + ms * 16 + col) * EMB
                               + h * HD2 + a * HD;
#pragma unroll
        for (int kk = 0; kk < 2; ++kk)
            qa[kk] = ld_frag(qrow + kk * 32 + kbase, qrow + kk * 32 + kbase + 16);
    }

    float mrow[8], lrow[8];
    v8f accO[8];
#pragma unroll
    for (int r = 0; r < 8; ++r) { mrow[r] = -3.0e38f; lrow[r] = 0.f; }
#pragma unroll
    for (int j = 0; j < 8; ++j)
#pragma unroll
        for (int e = 0; e < 8; ++e) accO[j][e] = 0.f;

    const int ldr = tid >> 2;           // 0..63: s-row for cooperative loads
    const int ldc = (tid & 3) * 32;     // channel part

#if defined(HAVE_TDM)
    // 64x128 bf16 tile: data_size=2B, pad every 64 DWORDs (one 256B row)
    // by 4 DWORDs (16B) -> LDS row stride 272B = 136 elements.
    const unsigned lds_off = (unsigned)(uintptr_t)(void*)Ks;
    v8i g1;
    g1[0] = (1 << 16) | (1 << 20) | (5 << 22) | (3 << 25);
    g1[1] = (int)(128u << 16);    // tensor_dim0 = 128
    g1[2] = (int)(64u  << 16);    // tensor_dim1 = 64
    g1[3] = (int)(128u << 16);    // tile_dim0 = 128
    g1[4] = 64;                   // tile_dim1 = 64
    g1[5] = EMB;                  // tensor_dim0_stride
    g1[6] = 0;
    g1[7] = 0;
#endif

    for (int s0 = 0; s0 < T_SEQ; s0 += 64) {
        // ---- stage K tile [64][128] ----
#if defined(HAVE_TDM)
        if (w == 0) {
            tdm_load_2d(lds_off,
                (unsigned long long)(uintptr_t)
                    (k + (size_t)(b * T_SEQ + s0) * EMB + h * HD2),
                g1);
            __builtin_amdgcn_s_wait_tensorcnt(0);
        }
#elif defined(HAVE_ASYNC)
        {
            const __bf16* kp = k + (size_t)(b * T_SEQ + s0 + ldr) * EMB + h * HD2 + ldc;
#pragma unroll
            for (int i = 0; i < 4; ++i)
                __builtin_amdgcn_global_load_async_to_lds_b128(
                    (const AS1 void*)(kp + i * 8),
                    (AS3 void*)&Ks[ldr * 136 + ldc + i * 8], 0, 0);
            __builtin_amdgcn_s_wait_asynccnt(0);
        }
#else
        {
            const __bf16* kp = k + (size_t)(b * T_SEQ + s0 + ldr) * EMB + h * HD2 + ldc;
#pragma unroll
            for (int i = 0; i < 4; ++i)
                *(v8bf*)&Ks[ldr * 136 + ldc + i * 8] = *(const v8bf*)(kp + i * 8);
        }
#endif
        // ---- stage V tile transposed [128][64] ----
        {
            const __bf16* vp = v + (size_t)(b * T_SEQ + s0 + ldr) * EMB + h * HD2 + ldc;
#pragma unroll
            for (int i = 0; i < 4; ++i) {
                v8bf vv = *(const v8bf*)(vp + i * 8);
#pragma unroll
                for (int e = 0; e < 8; ++e)
                    Vt[(ldc + i * 8 + e) * 72 + ldr] = vv[e];
            }
        }
        __syncthreads();

        // ---- logits: 16x64 tile = 4 N-frags x 2 K-steps of WMMA ----
        v8f sc[4];
#pragma unroll
        for (int j = 0; j < 4; ++j)
#pragma unroll
            for (int e = 0; e < 8; ++e) sc[j][e] = 0.f;
#pragma unroll
        for (int jn = 0; jn < 4; ++jn) {
            const int n0 = jn * 16 + col;
#pragma unroll
            for (int kk = 0; kk < 2; ++kk) {
                const int kb = a * 64 + kk * 32 + ((lane < 16) ? 0 : 8);
                v16bf bfr = ld_frag(&Ks[n0 * 136 + kb], &Ks[n0 * 136 + kb + 16]);
                sc[jn] = __builtin_amdgcn_wmma_f32_16x16x32_bf16(
                    false, qa[kk], false, bfr, (short)0, sc[jn], false, false);
            }
        }

        // ---- online softmax ----
#pragma unroll
        for (int r = 0; r < 8; ++r) {
            float lv[4];
#pragma unroll
            for (int jn = 0; jn < 4; ++jn) lv[jn] = sc[jn][r] * scale;
            float rmax = fmaxf(fmaxf(lv[0], lv[1]), fmaxf(lv[2], lv[3]));
#pragma unroll
            for (int m = 1; m < 16; m <<= 1) rmax = fmaxf(rmax, __shfl_xor(rmax, m, 16));
            const float newm = fmaxf(mrow[r], rmax);
            const float corr = __expf(mrow[r] - newm);
            float rs = 0.f;
#pragma unroll
            for (int jn = 0; jn < 4; ++jn) {
                const float pv = __expf(lv[jn] - newm);
                rs += pv;
                Ps[w * 1152 + (rOff + r) * 72 + jn * 16 + col] = (__bf16)pv;
            }
#pragma unroll
            for (int m = 1; m < 16; m <<= 1) rs += __shfl_xor(rs, m, 16);
            lrow[r] = lrow[r] * corr + rs;
            mrow[r] = newm;
#pragma unroll
            for (int jt = 0; jt < 8; ++jt) accO[jt][r] *= corr;
        }

        // ---- P A-fragments (same-wave LDS round trip; DS ops in-order) ----
        v16bf pa[2];
#pragma unroll
        for (int ps = 0; ps < 2; ++ps) {
            const __bf16* pp = &Ps[w * 1152 + col * 72 + ps * 32 + kbase];
            pa[ps] = ld_frag(pp, pp + 16);
        }
        // ---- O += P * V ----
#pragma unroll
        for (int jt = 0; jt < 8; ++jt) {
            const int c0 = jt * 16 + col;
#pragma unroll
            for (int ps = 0; ps < 2; ++ps) {
                const int kb = ps * 32 + ((lane < 16) ? 0 : 8);
                v16bf bfr = ld_frag(&Vt[c0 * 72 + kb], &Vt[c0 * 72 + kb + 16]);
                accO[jt] = __builtin_amdgcn_wmma_f32_16x16x32_bf16(
                    false, pa[ps], false, bfr, (short)0, accO[jt], false, false);
            }
        }
        __syncthreads();
    }

    // ---- finalize ----
    float oo[8][8];
#pragma unroll
    for (int jt = 0; jt < 8; ++jt)
#pragma unroll
        for (int r = 0; r < 8; ++r) oo[jt][r] = accO[jt][r] / lrow[r];

    if (a == 1) {
#pragma unroll
        for (int jt = 0; jt < 8; ++jt)
#pragma unroll
            for (int r = 0; r < 8; ++r)
                O2s[ms * 2048 + (rOff + r) * 128 + jt * 16 + col] = oo[jt][r];
    }
    __syncthreads();
    if (a == 0) {
        float sub[8];
#pragma unroll
        for (int jt = 0; jt < 8; ++jt) sub[jt] = subln[jt * 16 + col];
        float att[8][8], ssq[8];
#pragma unroll
        for (int r = 0; r < 8; ++r) ssq[r] = 0.f;
#pragma unroll
        for (int jt = 0; jt < 8; ++jt)
#pragma unroll
            for (int r = 0; r < 8; ++r) {
                const float t = oo[jt][r]
                    - lam * O2s[ms * 2048 + (rOff + r) * 128 + jt * 16 + col];
                att[jt][r] = t;
                ssq[r] += t * t;
            }
#pragma unroll
        for (int r = 0; r < 8; ++r) {
#pragma unroll
            for (int m = 1; m < 16; m <<= 1) ssq[r] += __shfl_xor(ssq[r], m, 16);
        }
#pragma unroll
        for (int r = 0; r < 8; ++r) {
            const float rms = rsqrtf(ssq[r] * (1.f / 128.f) + EPS_RMS);
            const size_t row = (size_t)(b * T_SEQ + m0 + ms * 16 + rOff + r);
#pragma unroll
            for (int jt = 0; jt < 8; ++jt) {
                const float vv = att[jt][r] * rms * sub[jt] * ONE_MINUS_LI;
                o[row * EMB + h * HD2 + jt * 16 + col] = (__bf16)vv;
            }
        }
    }
}

// ---------------------------------------------------------------------------
extern "C" void kernel_launch(void* const* d_in, const int* in_sizes, int n_in,
                              void* d_out, int out_size, void* d_ws, size_t ws_size,
                              hipStream_t stream) {
    const float* x    = (const float*)d_in[0];
    const float* Wq   = (const float*)d_in[1];
    const float* Wk   = (const float*)d_in[2];
    const float* Wv   = (const float*)d_in[3];
    const float* Wo   = (const float*)d_in[4];
    const float* lq1  = (const float*)d_in[5];
    const float* lk1  = (const float*)d_in[6];
    const float* lq2  = (const float*)d_in[7];
    const float* lk2  = (const float*)d_in[8];
    const float* subw = (const float*)d_in[9];

    const size_t XN = (size_t)4096 * 2048;           // B*T x E
    const size_t need = 4 * XN * sizeof(__bf16) + sizeof(float);
    if (ws_size < need) return;

    __bf16* qb    = (__bf16*)d_ws;
    __bf16* kb    = qb + XN;
    __bf16* vb    = kb + XN;
    __bf16* attnb = vb + XN;
    float*  lamp  = (float*)(attnb + XN);

    lambda_kernel<<<1, 32, 0, stream>>>(lq1, lk1, lq2, lk2, lamp);

    dim3 gg(2048 / 128, 4096 / 128);
    gemm_nt<float, __bf16><<<gg, 256, 0, stream>>>(x, Wq, qb, 4096, 2048, 2048);
    gemm_nt<float, __bf16><<<gg, 256, 0, stream>>>(x, Wk, kb, 4096, 2048, 2048);
    gemm_nt<float, __bf16><<<gg, 256, 0, stream>>>(x, Wv, vb, 4096, 2048, 2048);

    dim3 ga(T_SEQ / 64, NHEAD, 2);
    diff_attn<<<ga, 256, 0, stream>>>(qb, kb, vb, lamp, subw, attnb);

    gemm_nt<__bf16, float><<<gg, 256, 0, stream>>>(attnb, Wo, (float*)d_out,
                                                   4096, 2048, 2048);
}
